// CustomTransformerEncoderLayer_74328704024758
// MI455X (gfx1250) — matrix operand order repaired and checked by reading, not compiled, not run on hardware
//
#include <hip/hip_runtime.h>
#include <math.h>

// ---------------------------------------------------------------------------
// Types
// ---------------------------------------------------------------------------
typedef __attribute__((ext_vector_type(16))) __bf16  v16bf;
typedef __attribute__((ext_vector_type(8)))  float   v8f;
typedef __attribute__((ext_vector_type(4)))  unsigned int u32x4;
typedef __attribute__((ext_vector_type(2)))  unsigned int u32x2;
typedef __attribute__((ext_vector_type(4)))  float   f32x4;

union FragU { u32x4 q[2]; v16bf v; };
union Pack8 { u32x4 q; __bf16 b[8]; };
union Pack4 { u32x2 q; __bf16 b[4]; };

#define BDIM   8
#define NSEQ   1024
#define NHEADS 12
#define DMODEL 768
#define DK     64
#define DFF    3072
#define MROWS  (BDIM * NSEQ)          // 8192

// LDS row strides (in bf16 elements); 16B-aligned rows, conflict-reduced
#define STRIDE32 56                   // for BK=32 tiles (112 B rows)
#define STRIDE64 72                   // for K=64 tiles (144 B rows)

// ---------------------------------------------------------------------------
// CDNA5 async global->LDS copy (ASYNCcnt path). The low 32 bits of a flat
// __shared__ pointer are the wave-relative LDS byte offset (LDS aperture).
// ---------------------------------------------------------------------------
__device__ __forceinline__ void async_copy_b128(const __bf16* lds_dst,
                                                const void* gsrc) {
  unsigned lds_off = (unsigned)(unsigned long long)(uintptr_t)lds_dst;
  asm volatile("global_load_async_to_lds_b128 %0, %1, off"
               :: "v"(lds_off), "v"(gsrc)
               : "memory");
}
__device__ __forceinline__ void async_wait_all() {
  asm volatile("s_wait_asynccnt 0x0" ::: "memory");
}

// ---------------------------------------------------------------------------
// Fragment loaders (per documented CDNA5 16-bit WMMA VGPR layouts).
// A 16x32 bf16: lane-half h: elements 0..7 -> K=8h..8h+7, 8..15 -> K=16+8h..
// B 32x16 bf16 (staged transposed in LDS as Bt[n][k]): same K order as A.
// ---------------------------------------------------------------------------
__device__ __forceinline__ v16bf ld_afrag(const __bf16* base, int h) {
  FragU f;
  const u32x4* p = (const u32x4*)base;
  f.q[0] = p[h];          // bytes 16h      -> K = 8h .. 8h+7
  f.q[1] = p[2 + h];      // bytes 32+16h   -> K = 16+8h .. 16+8h+7
  return f.v;
}
__device__ __forceinline__ v16bf ld_bfrag(const __bf16* base, int h) {
  FragU f;
  const u32x4* p = (const u32x4*)base;
  f.q[0] = p[h];          // bytes 16h      -> K = 8h .. 8h+7
  f.q[1] = p[2 + h];      // bytes 32+16h   -> K = 16+8h .. 16+8h+7
  return f.v;
}

__device__ __forceinline__ v8f wmma_bf16(v16bf a, v16bf b, v8f c) {
  return __builtin_amdgcn_wmma_f32_16x16x32_bf16(
      /*neg_a=*/false, a, /*neg_b=*/false, b,
      /*c_mod=*/(short)0, c, /*reuse_a=*/false, /*reuse_b=*/false);
}

__device__ __forceinline__ float gelu_exact(float x) {
  return 0.5f * x * (1.0f + erff(x * 0.70710678118654752f));
}

// Deterministic stateless gumbel noise (splitmix64 hash of the flat index).
__device__ __forceinline__ float gumbel_noise(unsigned long long idx) {
  unsigned long long z = idx + 0x9E3779B97F4A7C15ull;
  z = (z ^ (z >> 30)) * 0xBF58476D1CE4E5B9ull;
  z = (z ^ (z >> 27)) * 0x94D049BB133111EBull;
  z ^= (z >> 31);
  float u = ((unsigned int)(z >> 40) + 0.5f) * (1.0f / 16777216.0f);
  return -logf(-logf(u));
}

// ---------------------------------------------------------------------------
// f32 -> bf16 conversion (grid-stride)
// ---------------------------------------------------------------------------
__global__ void f32_to_bf16_kernel(const float* __restrict__ in,
                                   __bf16* __restrict__ out, int n) {
  for (int i = blockIdx.x * blockDim.x + threadIdx.x; i < n;
       i += gridDim.x * blockDim.x)
    out[i] = (__bf16)in[i];
}

// ---------------------------------------------------------------------------
// Generic bf16 WMMA GEMM: C[M,N] = act(A[M,K] @ B[K,N] + bias)
// 128x128 tile per 256-thread (8-wave) workgroup; BK=32, LDS double-buffered.
// A tile staged with GLOBAL_LOAD_ASYNC_TO_LDS_B128 (ASYNCcnt); B tile staged
// synchronously (needs transpose into Bt[n][k]).
// Wave grid 4(M) x 2(N); wave tile 32x64 = 2x4 WMMA accumulators.
// ACT: 0=none, 1=exact GELU. WF/WB: write f32 / bf16 outputs.
// ---------------------------------------------------------------------------
template <int ACT, bool WF, bool WB>
__global__ __launch_bounds__(256)
void gemm_bf16_wmma(const __bf16* __restrict__ A, const __bf16* __restrict__ Bm,
                    const float* __restrict__ bias,
                    float* __restrict__ Cf, __bf16* __restrict__ Cb,
                    int M, int N, int K) {
  __shared__ __align__(16) __bf16 As[2][128 * STRIDE32];
  __shared__ __align__(16) __bf16 Bt[2][128 * STRIDE32];
  const int tid  = threadIdx.x;
  const int lane = tid & 31;
  const int wave = tid >> 5;
  const int wm   = wave >> 1;       // 0..3
  const int wn   = wave & 1;        // 0..1
  const int h    = lane >> 4;
  const int l16  = lane & 15;
  const int m0   = blockIdx.y * 128;
  const int n0   = blockIdx.x * 128;

  // Per-thread staging coordinates
  const int ar0  = (tid + 0)   >> 2, aseg0 = (tid + 0)   & 3;
  const int ar1  = (tid + 256) >> 2, aseg1 = (tid + 256) & 3;
  const int bk0  = (tid + 0)   >> 4, boct0 = (tid + 0)   & 15;
  const int bk1  = (tid + 256) >> 4, boct1 = (tid + 256) & 15;

  auto stage = [&](int buf, int kb) {
    // A tile 128x32: async straight copy, 2 x b128 per thread
    async_copy_b128(&As[buf][ar0 * STRIDE32 + aseg0 * 8],
                    A + (size_t)(m0 + ar0) * K + kb + aseg0 * 8);
    async_copy_b128(&As[buf][ar1 * STRIDE32 + aseg1 * 8],
                    A + (size_t)(m0 + ar1) * K + kb + aseg1 * 8);
    // B tile 32x128 transposed into Bt[n][k] (sync; scatter stores)
    Pack8 d0, d1;
    d0.q = *(const u32x4*)(Bm + (size_t)(kb + bk0) * N + n0 + boct0 * 8);
    d1.q = *(const u32x4*)(Bm + (size_t)(kb + bk1) * N + n0 + boct1 * 8);
#pragma unroll
    for (int i = 0; i < 8; ++i) Bt[buf][(boct0 * 8 + i) * STRIDE32 + bk0] = d0.b[i];
#pragma unroll
    for (int i = 0; i < 8; ++i) Bt[buf][(boct1 * 8 + i) * STRIDE32 + bk1] = d1.b[i];
  };

  v8f acc[2][4];
#pragma unroll
  for (int i = 0; i < 2; ++i)
#pragma unroll
    for (int j = 0; j < 4; ++j) acc[i][j] = (v8f){0, 0, 0, 0, 0, 0, 0, 0};

  const int nsteps = K >> 5;
  stage(0, 0);
  for (int s = 0; s < nsteps; ++s) {
    const int buf = s & 1;
    async_wait_all();      // this wave's async A loads for stage s are in LDS
    __syncthreads();       // all waves' A (async) + B (ds_store) stage s visible
    if (s + 1 < nsteps) stage(buf ^ 1, (s + 1) << 5);

    v16bf af[2], bfv[4];
#pragma unroll
    for (int i = 0; i < 2; ++i)
      af[i] = ld_afrag(&As[buf][(wm * 32 + i * 16 + l16) * STRIDE32], h);
#pragma unroll
    for (int j = 0; j < 4; ++j)
      bfv[j] = ld_bfrag(&Bt[buf][(wn * 64 + j * 16 + l16) * STRIDE32], h);
#pragma unroll
    for (int i = 0; i < 2; ++i)
#pragma unroll
      for (int j = 0; j < 4; ++j) acc[i][j] = wmma_bf16(af[i], bfv[j], acc[i][j]);
    __syncthreads();       // done reading buf before it is rewritten at s+2
  }

  // Epilogue: C/D layout -> lane = col%16 + 16*(row in upper half), vgpr -> row
#pragma unroll
  for (int i = 0; i < 2; ++i) {
#pragma unroll
    for (int j = 0; j < 4; ++j) {
      int col = n0 + wn * 64 + j * 16 + l16;
      float bv = bias[col];
#pragma unroll
      for (int v = 0; v < 8; ++v) {
        int row = m0 + wm * 32 + i * 16 + v + 8 * h;
        float x = acc[i][j][v] + bv;
        if constexpr (ACT == 1) x = gelu_exact(x);
        size_t o = (size_t)row * N + col;
        if constexpr (WF) Cf[o] = x;
        if constexpr (WB) Cb[o] = (__bf16)x;
      }
    }
  }
}

// ---------------------------------------------------------------------------
// Attention scores: S[bh] = Q[bh] @ K[bh]^T / sqrt(dk) + gumbel; per-(b,h)
// 128x128 output tile per workgroup, K=64 (two WMMA k-steps, fully staged).
// Q and K tiles are plain row copies -> async global->LDS.
// ---------------------------------------------------------------------------
__global__ __launch_bounds__(256)
void attn_scores_wmma(const __bf16* __restrict__ Q, const __bf16* __restrict__ Kmat,
                      float* __restrict__ Sout) {
  __shared__ __align__(16) __bf16 Qs[128 * STRIDE64];
  __shared__ __align__(16) __bf16 Ks[128 * STRIDE64];
  const int tid  = threadIdx.x;
  const int lane = tid & 31;
  const int wave = tid >> 5;
  const int wm   = wave >> 1, wn = wave & 1;
  const int h    = lane >> 4, l16 = lane & 15;
  const int m0   = blockIdx.y * 128, n0 = blockIdx.x * 128;
  const int bh   = blockIdx.z;
  const int b    = bh / NHEADS, hd = bh % NHEADS;
  const __bf16* Qb = Q    + (size_t)b * NSEQ * DMODEL + hd * DK;
  const __bf16* Kb = Kmat + (size_t)b * NSEQ * DMODEL + hd * DK;

#pragma unroll
  for (int it = 0; it < 4; ++it) {
    int idx = tid + it * 256;
    int r = idx >> 3, seg = idx & 7;
    async_copy_b128(&Qs[r * STRIDE64 + seg * 8],
                    Qb + (size_t)(m0 + r) * DMODEL + seg * 8);
    async_copy_b128(&Ks[r * STRIDE64 + seg * 8],
                    Kb + (size_t)(n0 + r) * DMODEL + seg * 8);
  }
  async_wait_all();
  __syncthreads();

  v8f acc[2][4];
#pragma unroll
  for (int i = 0; i < 2; ++i)
#pragma unroll
    for (int j = 0; j < 4; ++j) acc[i][j] = (v8f){0, 0, 0, 0, 0, 0, 0, 0};

#pragma unroll
  for (int ks = 0; ks < 2; ++ks) {
    int kb = ks * 32;
    v16bf af[2], bfv[4];
#pragma unroll
    for (int i = 0; i < 2; ++i)
      af[i] = ld_afrag(&Qs[(wm * 32 + i * 16 + l16) * STRIDE64 + kb], h);
#pragma unroll
    for (int j = 0; j < 4; ++j)
      bfv[j] = ld_bfrag(&Ks[(wn * 64 + j * 16 + l16) * STRIDE64 + kb], h);
#pragma unroll
    for (int i = 0; i < 2; ++i)
#pragma unroll
      for (int j = 0; j < 4; ++j) acc[i][j] = wmma_bf16(af[i], bfv[j], acc[i][j]);
  }

  const float scale = 0.125f;  // 1/sqrt(64)
#pragma unroll
  for (int i = 0; i < 2; ++i) {
#pragma unroll
    for (int j = 0; j < 4; ++j) {
      int col = n0 + wn * 64 + j * 16 + l16;
#pragma unroll
      for (int v = 0; v < 8; ++v) {
        int row = m0 + wm * 32 + i * 16 + v + 8 * h;
        size_t flat = ((size_t)bh * NSEQ + row) * NSEQ + col;
        Sout[flat] = acc[i][j][v] * scale + gumbel_noise(flat);
      }
    }
  }
}

// ---------------------------------------------------------------------------
// Row softmax (in place), one 256-thread block per 1024-wide row.
// ---------------------------------------------------------------------------
__global__ __launch_bounds__(256)
void softmax_rows(float* __restrict__ S) {
  __shared__ float red[256];
  const int tid = threadIdx.x;
  float* row = S + (size_t)blockIdx.x * NSEQ;
  float v[4];
  float m = -3.0e38f;
#pragma unroll
  for (int i = 0; i < 4; ++i) { v[i] = row[tid + i * 256]; m = fmaxf(m, v[i]); }
  red[tid] = m; __syncthreads();
  for (int st = 128; st > 0; st >>= 1) {
    if (tid < st) red[tid] = fmaxf(red[tid], red[tid + st]);
    __syncthreads();
  }
  m = red[0]; __syncthreads();
  float s = 0.f;
#pragma unroll
  for (int i = 0; i < 4; ++i) { v[i] = expf(v[i] - m); s += v[i]; }
  red[tid] = s; __syncthreads();
  for (int st = 128; st > 0; st >>= 1) {
    if (tid < st) red[tid] += red[tid + st];
    __syncthreads();
  }
  float inv = 1.0f / red[0];
#pragma unroll
  for (int i = 0; i < 4; ++i) row[tid + i * 256] = v[i] * inv;
}

// ---------------------------------------------------------------------------
// attn @ V per (b,h): [1024,1024] @ [1024,64] -> writes f32 [B,N,DMODEL] slice.
// 128(M) x 64(N) tile; wave grid 4x2, wave tile 32x32 = 2x2 accumulators.
// Attention probs converted f32 -> bf16 while staging A (sync path).
// ---------------------------------------------------------------------------
__global__ __launch_bounds__(256)
void attn_av_wmma(const float* __restrict__ Attn, const __bf16* __restrict__ V,
                  float* __restrict__ Out) {
  __shared__ __align__(16) __bf16 As[128 * STRIDE32];
  __shared__ __align__(16) __bf16 Bt[64 * STRIDE32];
  const int tid  = threadIdx.x;
  const int lane = tid & 31;
  const int wave = tid >> 5;
  const int wm   = wave >> 1, wn = wave & 1;
  const int h    = lane >> 4, l16 = lane & 15;
  const int m0   = blockIdx.x * 128;
  const int bh   = blockIdx.y;
  const int b    = bh / NHEADS, hd = bh % NHEADS;
  const float*  Ab = Attn + (size_t)bh * NSEQ * NSEQ;
  const __bf16* Vb = V + (size_t)b * NSEQ * DMODEL + hd * DK;

  v8f acc[2][2];
#pragma unroll
  for (int i = 0; i < 2; ++i)
#pragma unroll
    for (int j = 0; j < 2; ++j) acc[i][j] = (v8f){0, 0, 0, 0, 0, 0, 0, 0};

  for (int kb = 0; kb < NSEQ; kb += 32) {
    // Stage A (f32 attn -> bf16), 128x32
#pragma unroll
    for (int it = 0; it < 4; ++it) {
      int idx = tid + it * 256;
      int r = idx >> 3, seg = idx & 7;
      f32x4 d = *(const f32x4*)(Ab + (size_t)(m0 + r) * NSEQ + kb + seg * 4);
      Pack4 p;
#pragma unroll
      for (int i = 0; i < 4; ++i) p.b[i] = (__bf16)d[i];
      *(u32x2*)(&As[r * STRIDE32 + seg * 4]) = p.q;
    }
    // Stage V tile 32x64 transposed into Bt[n][k]
    {
      int k = tid >> 3, oct = tid & 7;
      Pack8 d;
      d.q = *(const u32x4*)(Vb + (size_t)(kb + k) * DMODEL + oct * 8);
#pragma unroll
      for (int i = 0; i < 8; ++i) Bt[(oct * 8 + i) * STRIDE32 + k] = d.b[i];
    }
    __syncthreads();

    v16bf af[2], bfv[2];
#pragma unroll
    for (int i = 0; i < 2; ++i)
      af[i] = ld_afrag(&As[(wm * 32 + i * 16 + l16) * STRIDE32], h);
#pragma unroll
    for (int j = 0; j < 2; ++j)
      bfv[j] = ld_bfrag(&Bt[(wn * 32 + j * 16 + l16) * STRIDE32], h);
#pragma unroll
    for (int i = 0; i < 2; ++i)
#pragma unroll
      for (int j = 0; j < 2; ++j) acc[i][j] = wmma_bf16(af[i], bfv[j], acc[i][j]);
    __syncthreads();
  }

#pragma unroll
  for (int i = 0; i < 2; ++i) {
#pragma unroll
    for (int j = 0; j < 2; ++j) {
      int col = wn * 32 + j * 16 + l16;  // 0..63 within head
#pragma unroll
      for (int v = 0; v < 8; ++v) {
        int row = m0 + wm * 32 + i * 16 + v + 8 * h;
        Out[((size_t)b * NSEQ + row) * DMODEL + hd * DK + col] = acc[i][j][v];
      }
    }
  }
}

// ---------------------------------------------------------------------------
// y = LayerNorm(a + b) * g + beta ; optional f32 + bf16 outputs.
// One 256-thread block per 768-wide row (3 elements / thread).
// ---------------------------------------------------------------------------
__global__ __launch_bounds__(256)
void add_layernorm(const float* __restrict__ Xa, const float* __restrict__ Xb,
                   const float* __restrict__ g, const float* __restrict__ be,
                   float* __restrict__ Yf, __bf16* __restrict__ Yb) {
  __shared__ float red[256];
  const int tid = threadIdx.x;
  const size_t row = blockIdx.x;
  const float* a = Xa + row * DMODEL;
  const float* c = Xb + row * DMODEL;
  float v[3];
  float s = 0.f;
#pragma unroll
  for (int i = 0; i < 3; ++i) {
    int col = tid + i * 256;
    v[i] = a[col] + c[col];
    s += v[i];
  }
  red[tid] = s; __syncthreads();
  for (int st = 128; st > 0; st >>= 1) {
    if (tid < st) red[tid] += red[tid + st];
    __syncthreads();
  }
  float mean = red[0] * (1.0f / DMODEL); __syncthreads();
  float q = 0.f;
#pragma unroll
  for (int i = 0; i < 3; ++i) { float d = v[i] - mean; q += d * d; }
  red[tid] = q; __syncthreads();
  for (int st = 128; st > 0; st >>= 1) {
    if (tid < st) red[tid] += red[tid + st];
    __syncthreads();
  }
  float inv = rsqrtf(red[0] * (1.0f / DMODEL) + 1e-5f);
#pragma unroll
  for (int i = 0; i < 3; ++i) {
    int col = tid + i * 256;
    float y = (v[i] - mean) * inv * g[col] + be[col];
    if (Yf) Yf[row * DMODEL + col] = y;
    if (Yb) Yb[row * DMODEL + col] = (__bf16)y;
  }
}

// ---------------------------------------------------------------------------
// Host launcher
// ---------------------------------------------------------------------------
extern "C" void kernel_launch(void* const* d_in, const int* in_sizes, int n_in,
                              void* d_out, int out_size, void* d_ws, size_t ws_size,
                              hipStream_t stream) {
  (void)in_sizes; (void)n_in; (void)out_size; (void)ws_size;
  const float* src = (const float*)d_in[0];
  const float* Wq  = (const float*)d_in[1];
  const float* bq  = (const float*)d_in[2];
  const float* Wk  = (const float*)d_in[3];
  const float* bk  = (const float*)d_in[4];
  const float* Wv  = (const float*)d_in[5];
  const float* bv  = (const float*)d_in[6];
  const float* W1  = (const float*)d_in[7];
  const float* b1  = (const float*)d_in[8];
  const float* W2  = (const float*)d_in[9];
  const float* b2  = (const float*)d_in[10];
  const float* g1  = (const float*)d_in[11];
  const float* be1 = (const float*)d_in[12];
  const float* g2  = (const float*)d_in[13];
  const float* be2 = (const float*)d_in[14];

  char* ws = (char*)d_ws;
  size_t off = 0;
  auto alloc = [&](size_t bytes) -> void* {
    void* p = ws + off;
    off += (bytes + 255) & ~(size_t)255;
    return p;
  };
  __bf16* srcb = (__bf16*)alloc((size_t)MROWS * DMODEL * 2);
  __bf16* Wqb  = (__bf16*)alloc((size_t)DMODEL * DMODEL * 2);
  __bf16* Wkb  = (__bf16*)alloc((size_t)DMODEL * DMODEL * 2);
  __bf16* Wvb  = (__bf16*)alloc((size_t)DMODEL * DMODEL * 2);
  __bf16* W1b  = (__bf16*)alloc((size_t)DMODEL * DFF * 2);
  __bf16* W2b  = (__bf16*)alloc((size_t)DFF * DMODEL * 2);
  __bf16* Qb   = (__bf16*)alloc((size_t)MROWS * DMODEL * 2);
  __bf16* Kb   = (__bf16*)alloc((size_t)MROWS * DMODEL * 2);
  __bf16* Vb   = (__bf16*)alloc((size_t)MROWS * DMODEL * 2);
  float*  attn_out = (float*)alloc((size_t)MROWS * DMODEL * 4);
  float*  xf   = (float*)alloc((size_t)MROWS * DMODEL * 4);
  __bf16* xb   = (__bf16*)alloc((size_t)MROWS * DMODEL * 2);
  __bf16* hb   = (__bf16*)alloc((size_t)MROWS * DFF * 2);
  float*  ff   = (float*)alloc((size_t)MROWS * DMODEL * 4);

  float* outp = (float*)d_out;                                  // [B,N,D]
  float* attn = (float*)d_out + (size_t)MROWS * DMODEL;         // [B,H,N,N]

  // 1) f32 -> bf16 conversions
  f32_to_bf16_kernel<<<512, 256, 0, stream>>>(src, srcb, MROWS * DMODEL);
  f32_to_bf16_kernel<<<256, 256, 0, stream>>>(Wq, Wqb, DMODEL * DMODEL);
  f32_to_bf16_kernel<<<256, 256, 0, stream>>>(Wk, Wkb, DMODEL * DMODEL);
  f32_to_bf16_kernel<<<256, 256, 0, stream>>>(Wv, Wvb, DMODEL * DMODEL);
  f32_to_bf16_kernel<<<512, 256, 0, stream>>>(W1, W1b, DMODEL * DFF);
  f32_to_bf16_kernel<<<512, 256, 0, stream>>>(W2, W2b, DFF * DMODEL);

  // 2) QKV projections (bf16 outputs)
  gemm_bf16_wmma<0, false, true><<<dim3(DMODEL / 128, MROWS / 128), 256, 0, stream>>>(
      srcb, Wqb, bq, nullptr, Qb, MROWS, DMODEL, DMODEL);
  gemm_bf16_wmma<0, false, true><<<dim3(DMODEL / 128, MROWS / 128), 256, 0, stream>>>(
      srcb, Wkb, bk, nullptr, Kb, MROWS, DMODEL, DMODEL);
  gemm_bf16_wmma<0, false, true><<<dim3(DMODEL / 128, MROWS / 128), 256, 0, stream>>>(
      srcb, Wvb, bv, nullptr, Vb, MROWS, DMODEL, DMODEL);

  // 3) scores + gumbel -> attn region of d_out (pre-softmax)
  attn_scores_wmma<<<dim3(NSEQ / 128, NSEQ / 128, BDIM * NHEADS), 256, 0, stream>>>(
      Qb, Kb, attn);

  // 4) softmax rows in place
  softmax_rows<<<BDIM * NHEADS * NSEQ, 256, 0, stream>>>(attn);

  // 5) attn @ V -> attn_out f32
  attn_av_wmma<<<dim3(NSEQ / 128, BDIM * NHEADS), 256, 0, stream>>>(
      attn, Vb, attn_out);

  // 6) x = LN(src + attn_out)
  add_layernorm<<<MROWS, 256, 0, stream>>>(src, attn_out, g1, be1, xf, xb);

  // 7) h = gelu(x @ W1 + b1) (bf16)
  gemm_bf16_wmma<1, false, true><<<dim3(DFF / 128, MROWS / 128), 256, 0, stream>>>(
      xb, W1b, b1, nullptr, hb, MROWS, DFF, DMODEL);

  // 8) ff = h @ W2 + b2 (f32)
  gemm_bf16_wmma<0, true, false><<<dim3(DMODEL / 128, MROWS / 128), 256, 0, stream>>>(
      hb, W2b, b2, ff, nullptr, MROWS, DMODEL, DFF);

  // 9) out = LN(x + ff)
  add_layernorm<<<MROWS, 256, 0, stream>>>(xf, ff, g2, be2, outp, nullptr);
}
